// Attention1_4801773437147
// MI455X (gfx1250) — matrix operand order repaired and checked by reading, not compiled, not run on hardware
//
#include <hip/hip_runtime.h>
#include <hip/hip_bf16.h>

// ---------------- problem constants ----------------
#define TN   16384   // sequence length
#define TD   512     // model dim
#define TQKV 1536    // 3*TD
#define H    8       // heads
#define DH   64      // dim head
#define LM   256     // landmarks
#define LG   64      // TN / LM  (landmark group)
#define SPLIT 16     // n-split for attn3@v flash pass

typedef __bf16 bhalf;
typedef __attribute__((ext_vector_type(16))) __bf16 v16bf;
typedef __attribute__((ext_vector_type(8)))  __bf16 v8bf;
typedef __attribute__((ext_vector_type(8)))  float  v8f;
typedef __attribute__((ext_vector_type(2)))  float  v2f;

#if __has_builtin(__builtin_amdgcn_wmma_f32_16x16x4_f32)
#define HAS_WMMA_F32 1
#else
#define HAS_WMMA_F32 0
#endif

// ---------------- WMMA helpers (CDNA5 wave32 layouts) ----------------
__device__ inline v8f wmma_bf(v16bf a, v16bf b, v8f c) {
  return __builtin_amdgcn_wmma_f32_16x16x32_bf16(false, a, false, b, (short)0, c, false, false);
}
__device__ inline v8f vzero8() {
  v8f z;
#pragma unroll
  for (int i = 0; i < 8; ++i) z[i] = 0.f;
  return z;
}
// A fragment, 16x32 bf16, source row-major (contraction along columns).
// lane<16: row=lane, K in {0..15}; lane>=16: row=lane-16, K in {16..31}
// -> one contiguous 32B read per lane
__device__ inline v16bf frag_a(const bhalf* base, int ld, int lane) {
  const bhalf* p = base + (size_t)(lane & 15) * ld;
  int kh = (lane >> 4) * 16;
  v16bf f;
#pragma unroll
  for (int i = 0; i < 16; ++i) f[i] = p[kh + i];
  return f;
}
// B fragment from K-contiguous storage (either Kmat^T with row-major [n][k],
// or a pre-transposed dense B). lane holds col n = lane&15; K contiguous per half
// -> one contiguous 32B read per lane
__device__ inline v16bf frag_bT(const bhalf* base, int ld, int lane) {
  const bhalf* p = base + (size_t)(lane & 15) * ld + (lane >> 4) * 16;
  v16bf f;
#pragma unroll
  for (int i = 0; i < 16; ++i) f[i] = p[i];
  return f;
}
// C/D layout: vgpr r, lane l -> (row = r + 8*(l>>4), col = l&15)

// ---------------- 1) LayerNorm -> bf16 ----------------
__global__ __launch_bounds__(128) void k_ln(const float* __restrict__ x,
                                            const float* __restrict__ gamma,
                                            const float* __restrict__ beta,
                                            bhalf* __restrict__ xln) {
  int n = blockIdx.x, t = threadIdx.x;
  const float4 d = ((const float4*)(x + (size_t)n * TD))[t];
  __shared__ float ssum[128], ssq[128];
  ssum[t] = d.x + d.y + d.z + d.w;
  ssq[t]  = d.x * d.x + d.y * d.y + d.z * d.z + d.w * d.w;
  __syncthreads();
  for (int s = 64; s > 0; s >>= 1) {
    if (t < s) { ssum[t] += ssum[t + s]; ssq[t] += ssq[t + s]; }
    __syncthreads();
  }
  float mu = ssum[0] * (1.0f / TD);
  float var = ssq[0] * (1.0f / TD) - mu * mu;
  float rs = rsqrtf(var + 1e-5f);
  int c = t * 4;
  bhalf* o = xln + (size_t)n * TD + c;
  o[0] = (bhalf)((d.x - mu) * rs * gamma[c + 0] + beta[c + 0]);
  o[1] = (bhalf)((d.y - mu) * rs * gamma[c + 1] + beta[c + 1]);
  o[2] = (bhalf)((d.z - mu) * rs * gamma[c + 2] + beta[c + 2]);
  o[3] = (bhalf)((d.w - mu) * rs * gamma[c + 3] + beta[c + 3]);
}

// ---------------- 2) Wqkv fp32 -> bf16, TRANSPOSED [TQKV][TD] ----------------
__global__ void k_cvtT(const float* __restrict__ w, bhalf* __restrict__ o) {
  int i = blockIdx.x * blockDim.x + threadIdx.x;  // over TD*TQKV, coalesced read
  if (i < TD * TQKV) {
    int k = i / TQKV, n = i % TQKV;
    o[(size_t)n * TD + k] = (bhalf)w[i];
  }
}

// ---------------- 3) w_eff = Wout @ Wdr ; b_eff ; zero scale slot ----------------
__global__ __launch_bounds__(512) void k_weff(const float* __restrict__ Wout,
                                              const float* __restrict__ Wdr,
                                              const float* __restrict__ bout,
                                              const float* __restrict__ bdr,
                                              float* weff, float* bscal, float* scalemax) {
  int t = threadIdx.x;
  float s = 0.f;
  for (int k = 0; k < TD; ++k) s += Wout[(size_t)t * TD + k] * Wdr[k];
  weff[t] = s;
  if (t == 0) {
    float b = 0.f;
    for (int k = 0; k < TD; ++k) b += bout[k] * Wdr[k];
    *bscal = b + bdr[0];
    *scalemax = 0.f;
  }
}

// ---------------- 4) QKV GEMM (bf16 WMMA, LDS-free: all frags contiguous global) ------
__global__ __launch_bounds__(256) void k_qkv(const bhalf* __restrict__ xln,
                                             const bhalf* __restrict__ wqbt,  // [TQKV][TD]
                                             bhalf* __restrict__ qb, bhalf* __restrict__ kb,
                                             float* __restrict__ vf, bhalf* __restrict__ vtb) {
  int t = threadIdx.x, lane = t & 31, w = t >> 5;
  int rb = blockIdx.y * 128, cb = blockIdx.x * 128;
  int wrow = (w & 3) * 32, wcol = (w >> 2) * 64;
  const bhalf* Arow0 = xln + (size_t)(rb + wrow) * TD;
  const bhalf* Brow0 = wqbt + (size_t)(cb + wcol) * TD;
  v8f acc[2][4];
#pragma unroll
  for (int mt = 0; mt < 2; ++mt)
#pragma unroll
    for (int nt = 0; nt < 4; ++nt) acc[mt][nt] = vzero8();

#pragma unroll 2
  for (int k0 = 0; k0 < TD; k0 += 32) {
    v16bf af[2], bfv[4];
#pragma unroll
    for (int mt = 0; mt < 2; ++mt) af[mt] = frag_a(Arow0 + (size_t)(mt * 16) * TD + k0, TD, lane);
#pragma unroll
    for (int nt = 0; nt < 4; ++nt) bfv[nt] = frag_bT(Brow0 + (size_t)(nt * 16) * TD + k0, TD, lane);
    if (k0 + 32 < TD) {  // global_prefetch_b8 next K-slab
      __builtin_prefetch(Arow0 + (size_t)(lane & 15) * TD + k0 + 32, 0, 3);
      __builtin_prefetch(Brow0 + (size_t)(lane & 15) * TD + k0 + 32, 0, 3);
    }
#pragma unroll
    for (int mt = 0; mt < 2; ++mt)
#pragma unroll
      for (int nt = 0; nt < 4; ++nt) acc[mt][nt] = wmma_bf(af[mt], bfv[nt], acc[mt][nt]);
  }
  int half = lane >> 4, lc = lane & 15;
#pragma unroll
  for (int mt = 0; mt < 2; ++mt)
#pragma unroll
    for (int nt = 0; nt < 4; ++nt)
#pragma unroll
      for (int r = 0; r < 8; ++r) {
        int gr = rb + wrow + mt * 16 + r + 8 * half;
        int gc = cb + wcol + nt * 16 + lc;
        float v = acc[mt][nt][r];
        int sec = gc >> 9, cc = gc & 511;
        int h = cc >> 6, d = cc & 63;
        size_t idx = ((size_t)h * TN + gr) * DH + d;
        if (sec == 0) qb[idx] = (bhalf)(v * 0.125f);   // dh^-0.5
        else if (sec == 1) kb[idx] = (bhalf)v;
        else {
          vf[idx] = v;
          vtb[((size_t)h * DH + d) * TN + gr] = (bhalf)v;  // transposed copy for P@V frags
        }
      }
}

// ---------------- 5) landmark means ----------------
__global__ __launch_bounds__(64) void k_landmarks(const bhalf* __restrict__ qb,
                                                  const bhalf* __restrict__ kb,
                                                  bhalf* __restrict__ ql, bhalf* __restrict__ kl) {
  int m = blockIdx.x, h = blockIdx.y, d = threadIdx.x;
  size_t base = ((size_t)h * TN + (size_t)m * LG) * DH + d;
  float sq = 0.f, sk = 0.f;
  for (int i = 0; i < LG; ++i) {
    sq += (float)qb[base + (size_t)i * DH];
    sk += (float)kb[base + (size_t)i * DH];
  }
  size_t o = ((size_t)h * LM + m) * DH + d;
  ql[o] = (bhalf)(sq * (1.0f / LG));
  kl[o] = (bhalf)(sk * (1.0f / LG));
}

// ---------------- 6) attn2 = softmax(q_l @ k_l^T), fp32 out ----------------
__global__ __launch_bounds__(32) void k_attn2(const bhalf* __restrict__ ql,
                                              const bhalf* __restrict__ kl,
                                              float* __restrict__ attn2) {
  int lane = threadIdx.x, h = blockIdx.y, mb = blockIdx.x * 16;
  const bhalf* qlh = ql + (size_t)h * LM * DH + (size_t)mb * DH;
  const bhalf* klh = kl + (size_t)h * LM * DH;
  v16bf aq[2];
#pragma unroll
  for (int ks = 0; ks < 2; ++ks) aq[ks] = frag_a(qlh + ks * 32, DH, lane);
  v8f s[16];
#pragma unroll
  for (int i = 0; i < 16; ++i) s[i] = vzero8();
  for (int ct = 0; ct < 16; ++ct)
#pragma unroll
    for (int ks = 0; ks < 2; ++ks)
      s[ct] = wmma_bf(aq[ks], frag_bT(klh + (size_t)(ct * 16) * DH + ks * 32, DH, lane), s[ct]);
  int half = lane >> 4, lc = lane & 15;
  float* out = attn2 + (size_t)h * LM * LM;
#pragma unroll
  for (int r = 0; r < 8; ++r) {
    int row = mb + r + 8 * half;
    float mx = -1e30f;
    for (int ct = 0; ct < 16; ++ct) mx = fmaxf(mx, s[ct][r]);
    for (int mm = 1; mm < 16; mm <<= 1) mx = fmaxf(mx, __shfl_xor(mx, mm, 32));
    float e[16], sum = 0.f;
    for (int ct = 0; ct < 16; ++ct) { e[ct] = __expf(s[ct][r] - mx); sum += e[ct]; }
    for (int mm = 1; mm < 16; mm <<= 1) sum += __shfl_xor(sum, mm, 32);
    float inv = 1.0f / sum;
    for (int ct = 0; ct < 16; ++ct) out[(size_t)row * LM + ct * 16 + lc] = e[ct] * inv;
  }
}

// ---------------- 7) pinv scale = max col-sum (row sums == 1 for softmax) ----------------
__global__ __launch_bounds__(256) void k_colsum(const float* __restrict__ attn2, float* scalemax) {
  int h = blockIdx.x, j = threadIdx.x;
  const float* a = attn2 + (size_t)h * LM * LM;
  float s = 0.f;
  for (int i = 0; i < LM; ++i) s += a[(size_t)i * LM + j];
  atomicMax((int*)scalemax, __float_as_int(s));  // positive floats: int order == float order
}

// ---------------- 8) z0 = attn2^T / scale ----------------
__global__ __launch_bounds__(256) void k_zinit(const float* __restrict__ attn2,
                                               const float* __restrict__ scalemax,
                                               float* __restrict__ z) {
  int i = blockIdx.x, h = blockIdx.y, j = threadIdx.x;
  float inv = 1.0f / *scalemax;
  z[((size_t)h * LM + i) * LM + j] = attn2[((size_t)h * LM + j) * LM + i] * inv;
}

// ---------------- 9) batched 256xK GEMM:  D = alpha*(A@B) + beta*A  (fp32 WMMA) -------
// K-loop fully unrolled: 16 slabs x 16 wmma = 256 static v_wmma_f32_16x16x4_f32
__global__ __launch_bounds__(128) void k_mm(const float* __restrict__ A, const float* __restrict__ B,
                                            float* __restrict__ D, int ncols, float alpha, float beta) {
  __shared__ float As[64][16];
  __shared__ float Bst[64][16];  // transposed B tile: [n][k] -> contiguous v2f operand loads
  int t = threadIdx.x, lane = t & 31, w = t >> 5;
  int half = lane >> 4, lc = lane & 15;
  int h = blockIdx.z, rb = blockIdx.y * 64, cb = blockIdx.x * 64;
  const float* Ah = A + (size_t)h * LM * LM;
  const float* Bh = B + (size_t)h * LM * ncols;
  float* Dh = D + (size_t)h * LM * ncols;
  int wrow = (w & 1) * 32, wcol = (w >> 1) * 32;
  v8f acc[2][2];
#pragma unroll
  for (int mt = 0; mt < 2; ++mt)
#pragma unroll
    for (int nt = 0; nt < 2; ++nt) acc[mt][nt] = vzero8();

#pragma unroll
  for (int k0 = 0; k0 < LM; k0 += 16) {
    int ar = t >> 1, ac = (t & 1) * 8;
    *(float4*)&As[ar][ac]     = *(const float4*)&Ah[(size_t)(rb + ar) * LM + k0 + ac];
    *(float4*)&As[ar][ac + 4] = *(const float4*)&Ah[(size_t)(rb + ar) * LM + k0 + ac + 4];
    int br = t >> 3, bc = (t & 7) * 8;
    float4 b0 = *(const float4*)&Bh[(size_t)(k0 + br) * ncols + cb + bc];
    float4 b1 = *(const float4*)&Bh[(size_t)(k0 + br) * ncols + cb + bc + 4];
    Bst[bc + 0][br] = b0.x; Bst[bc + 1][br] = b0.y; Bst[bc + 2][br] = b0.z; Bst[bc + 3][br] = b0.w;
    Bst[bc + 4][br] = b1.x; Bst[bc + 5][br] = b1.y; Bst[bc + 6][br] = b1.z; Bst[bc + 7][br] = b1.w;
    __syncthreads();
#if HAS_WMMA_F32
#pragma unroll
    for (int kk = 0; kk < 16; kk += 4) {
#pragma unroll
      for (int mt = 0; mt < 2; ++mt) {
        int arow = wrow + mt * 16 + lc;
        v2f av; av[0] = As[arow][kk + half * 2]; av[1] = As[arow][kk + half * 2 + 1];
#pragma unroll
        for (int nt = 0; nt < 2; ++nt) {
          int bcol = wcol + nt * 16 + lc;
          v2f bv; bv[0] = Bst[bcol][kk + half * 2]; bv[1] = Bst[bcol][kk + half * 2 + 1];
          acc[mt][nt] = __builtin_amdgcn_wmma_f32_16x16x4_f32(false, av, false, bv,
                                                              (short)0, acc[mt][nt], false, false);
        }
      }
    }
#else
#pragma unroll
    for (int mt = 0; mt < 2; ++mt)
#pragma unroll
      for (int nt = 0; nt < 2; ++nt)
#pragma unroll
        for (int r = 0; r < 8; ++r) {
          int row = wrow + mt * 16 + r + 8 * half, col = wcol + nt * 16 + lc;
          float a = 0.f;
          for (int kk = 0; kk < 16; ++kk) a += As[row][kk] * Bst[col][kk];
          acc[mt][nt][r] += a;
        }
#endif
    __syncthreads();
  }
#pragma unroll
  for (int mt = 0; mt < 2; ++mt)
#pragma unroll
    for (int nt = 0; nt < 2; ++nt)
#pragma unroll
      for (int r = 0; r < 8; ++r) {
        int row = rb + wrow + mt * 16 + r + 8 * half;
        int col = cb + wcol + nt * 16 + lc;
        Dh[(size_t)row * ncols + col] = alpha * acc[mt][nt][r] + beta * Ah[(size_t)row * LM + col];
      }
}

// ---------------- 10) flash-style attn3 @ v, n-split partials (LDS only for P bounce) --
__global__ __launch_bounds__(32) void k_a3v(const bhalf* __restrict__ ql,
                                            const bhalf* __restrict__ kb,
                                            const bhalf* __restrict__ vtb,  // [h][d][n]
                                            float* __restrict__ pO, float* __restrict__ pM,
                                            float* __restrict__ pL) {
  __shared__ bhalf pt[16][32];
  int lane = threadIdx.x;
  int mb = blockIdx.x * 16, h = blockIdx.y, zs = blockIdx.z;
  int half = lane >> 4, lc = lane & 15;
  const bhalf* qlh = ql + ((size_t)h * LM + mb) * DH;
  const bhalf* kh = kb + (size_t)h * TN * DH;
  const bhalf* vth = vtb + (size_t)h * DH * TN;
  v16bf aq[2];
#pragma unroll
  for (int ks = 0; ks < 2; ++ks) aq[ks] = frag_a(qlh + ks * 32, DH, lane);
  v8f O[4];
#pragma unroll
  for (int i = 0; i < 4; ++i) O[i] = vzero8();
  float mrow[8], lrow[8];
#pragma unroll
  for (int r = 0; r < 8; ++r) { mrow[r] = -1e30f; lrow[r] = 0.f; }

  int nbeg = zs * (TN / SPLIT), nend = nbeg + TN / SPLIT;
  for (int n0 = nbeg; n0 < nend; n0 += 32) {
    v8f s0 = vzero8(), s1 = vzero8();
#pragma unroll
    for (int ks = 0; ks < 2; ++ks) {
      s0 = wmma_bf(aq[ks], frag_bT(kh + (size_t)n0 * DH + ks * 32, DH, lane), s0);
      s1 = wmma_bf(aq[ks], frag_bT(kh + (size_t)(n0 + 16) * DH + ks * 32, DH, lane), s1);
    }
    if (n0 + 32 < nend)
      __builtin_prefetch(kh + (size_t)(n0 + 32 + lane) * DH, 0, 3);
#pragma unroll
    for (int r = 0; r < 8; ++r) {
      float mx = fmaxf(s0[r], s1[r]);
      for (int mm = 1; mm < 16; mm <<= 1) mx = fmaxf(mx, __shfl_xor(mx, mm, 32));
      float mnew = fmaxf(mrow[r], mx);
      float alpha = __expf(mrow[r] - mnew);
      float p0 = __expf(s0[r] - mnew), p1 = __expf(s1[r] - mnew);
      float ls = p0 + p1;
      for (int mm = 1; mm < 16; mm <<= 1) ls += __shfl_xor(ls, mm, 32);
      lrow[r] = lrow[r] * alpha + ls;
      mrow[r] = mnew;
#pragma unroll
      for (int ct = 0; ct < 4; ++ct) O[ct][r] *= alpha;
      int prow = r + 8 * half;
      pt[prow][lc] = (bhalf)p0;
      pt[prow][16 + lc] = (bhalf)p1;
    }
    __syncthreads();
    v16bf pa = frag_a(&pt[0][0], 32, lane);
#pragma unroll
    for (int ct = 0; ct < 4; ++ct)
      O[ct] = wmma_bf(pa, frag_bT(vth + (size_t)(ct * 16) * TN + n0, TN, lane), O[ct]);
    __syncthreads();
  }
  size_t pbase = (((size_t)zs * H + h) * LM + mb);
#pragma unroll
  for (int r = 0; r < 8; ++r) {
    int row = r + 8 * half;
#pragma unroll
    for (int ct = 0; ct < 4; ++ct) pO[(pbase + row) * DH + ct * 16 + lc] = O[ct][r];
    if (lc == 0) { pM[pbase + row] = mrow[r]; pL[pbase + row] = lrow[r]; }
  }
}

__global__ __launch_bounds__(256) void k_a3v_merge(const float* __restrict__ pO,
                                                   const float* __restrict__ pM,
                                                   const float* __restrict__ pL,
                                                   float* __restrict__ a3v) {
  int idx = blockIdx.x * blockDim.x + threadIdx.x;
  if (idx >= H * LM * DH) return;
  int d = idx & 63, hm = idx >> 6;
  float M = -1e30f;
  for (int z = 0; z < SPLIT; ++z) M = fmaxf(M, pM[(size_t)z * H * LM + hm]);
  float L = 0.f, o = 0.f;
  for (int z = 0; z < SPLIT; ++z) {
    float wgt = __expf(pM[(size_t)z * H * LM + hm] - M);
    L += wgt * pL[(size_t)z * H * LM + hm];
    o += wgt * pO[((size_t)z * H * LM + hm) * DH + d];
  }
  a3v[(size_t)hm * DH + d] = o / L;
}

// ---------------- 10b) ZW -> transposed bf16 [h][d][m] ----------------
__global__ void k_zwt(const float* __restrict__ zw, bhalf* __restrict__ o) {
  int i = blockIdx.x * blockDim.x + threadIdx.x;  // H*LM*DH
  if (i < H * LM * DH) {
    int d = i & 63, m = (i >> 6) & (LM - 1), h = i >> 14;
    o[((size_t)h * DH + d) * LM + m] = (bhalf)zw[i];
  }
}

// ---------------- 11) fused attn1 softmax + (attn1 @ ZW) ----------------
__global__ __launch_bounds__(128) void k_attn1(const bhalf* __restrict__ qb,
                                               const bhalf* __restrict__ kl,
                                               const bhalf* __restrict__ zwt,  // [h][d][m]
                                               float* __restrict__ aout) {
  __shared__ bhalf plds[4][16][LM];  // per-wave P tiles (32 KB)
  int t = threadIdx.x, lane = t & 31, w = t >> 5;
  int h = blockIdx.y, rb = blockIdx.x * 64 + w * 16;
  const bhalf* klh = kl + (size_t)h * LM * DH;
  const bhalf* zwh = zwt + (size_t)h * DH * LM;
  const bhalf* qrow = qb + ((size_t)h * TN + rb) * DH;
  v16bf aq[2];
#pragma unroll
  for (int ks = 0; ks < 2; ++ks) aq[ks] = frag_a(qrow + ks * 32, DH, lane);
  v8f s[16];
#pragma unroll
  for (int i = 0; i < 16; ++i) s[i] = vzero8();
  for (int ct = 0; ct < 16; ++ct)
#pragma unroll
    for (int ks = 0; ks < 2; ++ks)
      s[ct] = wmma_bf(aq[ks], frag_bT(klh + (size_t)(ct * 16) * DH + ks * 32, DH, lane), s[ct]);
  int half = lane >> 4, lc = lane & 15;
  float linv[8];
#pragma unroll
  for (int r = 0; r < 8; ++r) {
    float mx = -1e30f;
    for (int ct = 0; ct < 16; ++ct) mx = fmaxf(mx, s[ct][r]);
    for (int mm = 1; mm < 16; mm <<= 1) mx = fmaxf(mx, __shfl_xor(mx, mm, 32));
    float sum = 0.f;
    int prow = r + 8 * half;
    for (int ct = 0; ct < 16; ++ct) {
      float e = __expf(s[ct][r] - mx);
      sum += e;
      plds[w][prow][ct * 16 + lc] = (bhalf)e;
    }
    for (int mm = 1; mm < 16; mm <<= 1) sum += __shfl_xor(sum, mm, 32);
    linv[r] = 1.0f / sum;
  }
  __syncthreads();
  v8f O[4];
#pragma unroll
  for (int i = 0; i < 4; ++i) O[i] = vzero8();
  for (int k8 = 0; k8 < 8; ++k8) {
    v16bf pa = frag_a(&plds[w][0][0] + k8 * 32, LM, lane);
#pragma unroll
    for (int ct = 0; ct < 4; ++ct)
      O[ct] = wmma_bf(pa, frag_bT(zwh + (size_t)(ct * 16) * LM + k8 * 32, LM, lane), O[ct]);
  }
#pragma unroll
  for (int r = 0; r < 8; ++r) {
    int gr = rb + r + 8 * half;
#pragma unroll
    for (int ct = 0; ct < 4; ++ct)
      aout[(size_t)gr * (H * DH) + h * DH + ct * 16 + lc] = O[ct][r] * linv[r];
  }
}

// ---------------- 12) depthwise conv residual + collapsed projection + final dot ------
__global__ __launch_bounds__(128) void k_final(const float* __restrict__ aout,
                                               const float* __restrict__ vf,
                                               const float* __restrict__ resw,
                                               const float* __restrict__ weff,
                                               const float* __restrict__ x,
                                               const float* __restrict__ Wdr,
                                               const float* __restrict__ bscal,
                                               float* __restrict__ out) {
  int n = blockIdx.x, t = threadIdx.x;
  float part = 0.f;
#pragma unroll
  for (int j = 0; j < 4; ++j) {
    int c = t * 4 + j;
    int h = c >> 6, d = c & 63;
    float conv = 0.f;
#pragma unroll
    for (int k = 0; k < 33; ++k) {
      int nn = n + k - 16;
      if (nn >= 0 && nn < TN) conv += resw[h * 33 + k] * vf[((size_t)h * TN + nn) * DH + d];
    }
    part += (aout[(size_t)n * TD + c] + conv) * weff[c];
    part += x[(size_t)n * TD + c] * Wdr[c];
  }
  __shared__ float red[128];
  red[t] = part;
  __syncthreads();
  for (int s = 64; s > 0; s >>= 1) {
    if (t < s) red[t] += red[t + s];
    __syncthreads();
  }
  if (t == 0) out[n] = red[0] + *bscal;
}

// ---------------- host launcher ----------------
extern "C" void kernel_launch(void* const* d_in, const int* in_sizes, int n_in,
                              void* d_out, int out_size, void* d_ws, size_t ws_size,
                              hipStream_t stream) {
  const float* x     = (const float*)d_in[0];
  const float* gamma = (const float*)d_in[1];
  const float* beta  = (const float*)d_in[2];
  const float* Wqkv  = (const float*)d_in[3];
  const float* resw  = (const float*)d_in[4];
  const float* Wout  = (const float*)d_in[5];
  const float* bout  = (const float*)d_in[6];
  const float* Wdr   = (const float*)d_in[7];
  const float* bdr   = (const float*)d_in[8];
  float* out = (float*)d_out;
  (void)in_sizes; (void)n_in; (void)out_size; (void)ws_size;

  size_t off = 0;
  char* base = (char*)d_ws;
  auto alloc = [&](size_t bytes) -> void* {
    off = (off + 255) & ~(size_t)255;
    void* p = base + off;
    off += bytes;
    return p;
  };
  bhalf* xln  = (bhalf*)alloc((size_t)TN * TD * 2);
  bhalf* wqbt = (bhalf*)alloc((size_t)TD * TQKV * 2);
  bhalf* qb   = (bhalf*)alloc((size_t)H * TN * DH * 2);
  bhalf* kb   = (bhalf*)alloc((size_t)H * TN * DH * 2);
  float* vf   = (float*)alloc((size_t)H * TN * DH * 4);
  bhalf* vtb  = (bhalf*)alloc((size_t)H * TN * DH * 2);
  bhalf* ql   = (bhalf*)alloc((size_t)H * LM * DH * 2);
  bhalf* kl   = (bhalf*)alloc((size_t)H * LM * DH * 2);
  float* attn2 = (float*)alloc((size_t)H * LM * LM * 4);
  float* z0  = (float*)alloc((size_t)H * LM * LM * 4);
  float* z1  = (float*)alloc((size_t)H * LM * LM * 4);
  float* txz = (float*)alloc((size_t)H * LM * LM * 4);
  float* tu1 = (float*)alloc((size_t)H * LM * LM * 4);
  float* tu2 = (float*)alloc((size_t)H * LM * LM * 4);
  float* a3v = (float*)alloc((size_t)H * LM * DH * 4);
  float* zwf = (float*)alloc((size_t)H * LM * DH * 4);
  bhalf* zwtb = (bhalf*)alloc((size_t)H * LM * DH * 2);
  float* aout = (float*)alloc((size_t)TN * TD * 4);
  float* weff = (float*)alloc(TD * 4);
  float* bscal = (float*)alloc(4);
  float* scalemax = (float*)alloc(4);
  float* pO = (float*)alloc((size_t)SPLIT * H * LM * DH * 4);
  float* pM = (float*)alloc((size_t)SPLIT * H * LM * 4);
  float* pL = (float*)alloc((size_t)SPLIT * H * LM * 4);

  k_ln<<<TN, 128, 0, stream>>>(x, gamma, beta, xln);
  k_cvtT<<<(TD * TQKV + 255) / 256, 256, 0, stream>>>(Wqkv, wqbt);
  k_weff<<<1, 512, 0, stream>>>(Wout, Wdr, bout, bdr, weff, bscal, scalemax);
  k_qkv<<<dim3(TQKV / 128, TN / 128), 256, 0, stream>>>(xln, wqbt, qb, kb, vf, vtb);
  k_landmarks<<<dim3(LM, H), 64, 0, stream>>>(qb, kb, ql, kl);
  k_attn2<<<dim3(LM / 16, H), 32, 0, stream>>>(ql, kl, attn2);
  k_colsum<<<H, 256, 0, stream>>>(attn2, scalemax);
  k_zinit<<<dim3(LM, H), 256, 0, stream>>>(attn2, scalemax, z0);

  auto MM = [&](const float* A, const float* B, float* D, int ncols, float alpha, float beta_) {
    k_mm<<<dim3(ncols / 64, LM / 64, H), 128, 0, stream>>>(A, B, D, ncols, alpha, beta_);
  };
  for (int it = 0; it < 6; ++it) {
    float* zin = (it & 1) ? z1 : z0;
    float* zout = (it & 1) ? z0 : z1;
    MM(attn2, zin, txz, 1.f, 0.f, LM);     // XZ = X @ Z
    MM(txz, txz, tu1, -1.f, 7.f, LM);      // U1 = 7*XZ - XZ@XZ
    MM(txz, tu1, tu2, -1.f, 15.f, LM);     // U2 = 15*XZ - XZ@U1
    MM(zin, tu2, zout, -0.25f, 3.25f, LM); // Z' = 3.25*Z - 0.25*Z@U2
  }
  // after 6 iters final Z is in z0
  k_a3v<<<dim3(LM / 16, H, SPLIT), 32, 0, stream>>>(ql, kb, vtb, pO, pM, pL);
  k_a3v_merge<<<(H * LM * DH + 255) / 256, 256, 0, stream>>>(pO, pM, pL, a3v);
  MM(z0, a3v, zwf, DH, 1.f, 0.f);          // ZW = Z @ (attn3@v)
  k_zwt<<<(H * LM * DH + 255) / 256, 256, 0, stream>>>(zwf, zwtb);
  k_attn1<<<dim3(TN / 64, H), 128, 0, stream>>>(qb, kl, zwtb, aout);
  k_final<<<TN, 128, 0, stream>>>(aout, vf, resw, weff, x, Wdr, bscal, out);
}
